// RiemannianManifold_12412455485913
// MI455X (gfx1250) — matrix operand order repaired and checked by reading, not compiled, not run on hardware
//
#include <hip/hip_runtime.h>
#include <math.h>

typedef __attribute__((ext_vector_type(16))) _Float16 v16h;
typedef __attribute__((ext_vector_type(8)))  _Float16 v8h;
typedef __attribute__((ext_vector_type(8)))  float    v8f;
typedef __attribute__((ext_vector_type(4)))  float    v4f;

namespace {
constexpr int kB  = 8;
constexpr int kN  = 256;
constexpr int kM  = 256;
constexpr int kD  = 16;    // inner dim D
constexpr int kH  = 64;    // hidden H
constexpr int kDD = 256;   // D*D
constexpr int kWaves  = 8;             // waves per block (8 * 32 = 256 threads)
constexpr int kMT     = kM / 16;       // 16 m-tiles per (b,n)
constexpr int kGroups = kB * kN * kMT; // 32768 point-groups of 16
constexpr int kGPW    = 4;             // point-groups (m-tiles) per wave
constexpr int kWavesTotal = kGroups / kGPW;  // 8192
}

__global__ __launch_bounds__(kWaves * 32)
void rmfd_wmma_kernel(const float* __restrict__ x1,
                      const float* __restrict__ x2,
                      const float* __restrict__ W1,
                      const float* __restrict__ b1,
                      const float* __restrict__ W2,
                      const float* __restrict__ b2,
                      float* __restrict__ out)
{
  // W2 transposed to f16: sW2T[n][k], n in [0,256), k in [0,64)  -> 32 KB
  __shared__ __align__(32) _Float16 sW2T[kDD * kH];
  // per-wave SiLU(hidden), row-major so GEMM2 A-operands are contiguous v16h
  __shared__ __align__(32) _Float16 sH[kWaves][16 * kH];
  // per-wave dx tile: 16 points x 16 f32
  __shared__ __align__(16) float    sDX[kWaves][16 * kD];

  const int tid  = threadIdx.x;
  const int wv   = tid >> 5;
  const int lane = tid & 31;
  const int p    = lane & 15;  // N-column within a WMMA tile
  const int hi   = lane >> 4;  // half-wave

  // ---- stage W2^T into LDS as f16 (float4 coalesced reads, 16 iters) ----
  for (int idx4 = tid * 4; idx4 < kH * kDD; idx4 += kWaves * 32 * 4) {
    const int k = idx4 >> 8;    // row of W2 (K)
    const int n = idx4 & 255;   // col of W2 (output unit)
    const v4f w4 = *(const v4f*)&W2[idx4];
    #pragma unroll
    for (int i = 0; i < 4; ++i) sW2T[(n + i) * kH + k] = (_Float16)w4[i];
  }
  __syncthreads();

  const int wave_g = blockIdx.x * kWaves + wv;
  if (wave_g >= kWavesTotal) return;
  const int gi0  = wave_g * kGPW;          // 4 consecutive m-tiles, same (b,n)
  const int mt0  = gi0 % kMT;
  const int nrow = (gi0 / kMT) % kN;
  const int bidx = gi0 / (kMT * kN);

  // ---- hoisted per-wave: W1 as WMMA B-operands (K=16, zero-pad to 32) ----
  // B layout (16-bit, 32x16): lane element j<8 holds B[K = 8*hi + j][N = p]; j>=8 -> K+16 (zero pad)
  v16h bw1[4];
  #pragma unroll
  for (int t = 0; t < 4; ++t) {
    #pragma unroll
    for (int j = 0; j < 16; ++j) {
      float w = 0.0f;
      if (j < 8) w = W1[(8 * hi + j) * kH + t * 16 + p];
      bw1[t][j] = (_Float16)w;
    }
  }
  float b1r[4];
  #pragma unroll
  for (int t = 0; t < 4; ++t) b1r[t] = b1[t * 16 + p];
  float b2r[16];
  #pragma unroll
  for (int e = 0; e < 16; ++e) b2r[e] = b2[e * 16 + p];

  // ---- hoisted: x1 row (constant across this wave's m-tiles) ----
  const int k0 = hi * 8;
  const float* x1p = x1 + ((size_t)(bidx * kN + nrow)) * kD + k0;   // (B,N,1,D)
  const v4f x1lo = *(const v4f*)x1p;
  const v4f x1hi = *(const v4f*)(x1p + 4);

  const v8f zf = {0.f, 0.f, 0.f, 0.f, 0.f, 0.f, 0.f, 0.f};
  const size_t outrow = ((size_t)(bidx * kN + nrow)) * kM;

  // ==== group loop: 4 m-tiles per wave (rolled to amortize setup) ====
  #pragma unroll 1
  for (int q = 0; q < kGPW; ++q) {
    const int m0 = (mt0 + q) * 16;

    // ---- build mid A-operand + dx ----
    // A layout: lane element j<8 -> A[M=p][K = 8*hi + j]; j>=8 -> K+16 (zero pad)
    const float* x2p = x2 + ((size_t)(bidx * kM + m0 + p)) * kD + k0; // (B,1,M,D)
    const v4f x2lo = *(const v4f*)x2p;
    const v4f x2hi = *(const v4f*)(x2p + 4);

    v16h a1;
    v4f dxlo, dxhi;
    #pragma unroll
    for (int j = 0; j < 4; ++j) {
      a1[j]      = (_Float16)(0.5f * (x1lo[j] + x2lo[j]));
      a1[4 + j]  = (_Float16)(0.5f * (x1hi[j] + x2hi[j]));
      a1[8 + j]  = (_Float16)0.0f;
      a1[12 + j] = (_Float16)0.0f;
      dxlo[j] = x2lo[j] - x1lo[j];
      dxhi[j] = x2hi[j] - x1hi[j];
    }
    *(v4f*)&sDX[wv][p * kD + k0]     = dxlo;
    *(v4f*)&sDX[wv][p * kD + k0 + 4] = dxhi;

    // ---- GEMM1: H = SiLU(mid @ W1 + b1); store row-major f16 H to LDS ----
    // layout: H[row][k] at row*64 + k
    #pragma unroll
    for (int t = 0; t < 4; ++t) {
      v8f hacc = __builtin_amdgcn_wmma_f32_16x16x32_f16(
          false, a1, false, bw1[t], (short)0, zf, false, false);
      const int coff = t * 16 + p;
      #pragma unroll
      for (int r = 0; r < 8; ++r) {
        const float hx = hacc[r] + b1r[t];
        const float s  = hx * __builtin_amdgcn_rcpf(1.0f + __expf(-hx)); // SiLU
        sH[wv][(r + 8 * hi) * kH + coff] = (_Float16)s;
      }
    }
    // same-wave LDS RAW fence (H + dx stores -> reads below)
    asm volatile("s_wait_dscnt 0" ::: "memory");

    // ---- reload H as GEMM2 A-operands: contiguous v16h per (lane, K-half) ----
    v16h a2[2];
    #pragma unroll
    for (int w = 0; w < 2; ++w)
      a2[w] = *(const v16h*)&sH[wv][p * kH + w * 32 + hi * 16];

    // ---- GEMM2 + fused epilogue: tile e == row e of L ----
    v8f y = zf;          // y[r] = (L^T dx)[d=p] for point r+8*hi (off-diag part)
    v8f diag = zf;       // captured raw diagonal (+bias) for d=p
    #pragma unroll
    for (int e = 0; e < 16; ++e) {
      const int nb = (e * 16 + p) * kH;   // W2T row for output col 16e+p
      const v16h bop0 = *(const v16h*)&sW2T[nb + 16 * hi];        // K 0..31
      const v16h bop1 = *(const v16h*)&sW2T[nb + 32 + 16 * hi];   // K 32..63
      v8f c = __builtin_amdgcn_wmma_f32_16x16x32_f16(
          false, a2[0], false, bop0, (short)0, zf, false, false);
      c = __builtin_amdgcn_wmma_f32_16x16x32_f16(
          false, a2[1], false, bop1, (short)0, c, false, false);

      const float bc = b2r[e];
      #pragma unroll
      for (int r = 0; r < 8; ++r) {
        const float v   = c[r] + bc;                        // raw[e*16+d] + b2
        const float dxv = sDX[wv][(r + 8 * hi) * kD + e];   // dx[point][e]
        y[r] += ((p < e) ? v : 0.0f) * dxv;                 // strict lower tri
        diag[r] = (p == e) ? v : diag[r];                   // capture diag (no trans)
      }
    }

    // ---- diagonal: softplus(diag + 1) * dx[point][p] — all lanes useful ----
    #pragma unroll
    for (int r = 0; r < 8; ++r) {
      const float td = diag[r] + 1.0f;                      // raw + identity
      const float sp = (td > 20.0f) ? td : __logf(1.0f + __expf(td));
      y[r] += sp * sDX[wv][(r + 8 * hi) * kD + p];
    }

    // ---- dist = sqrt(max(sum_d y^2, 1e-6)) ----
    float keep = 0.0f;
    #pragma unroll
    for (int r = 0; r < 8; ++r) {
      float s = y[r] * y[r];
      s += __shfl_xor(s, 1, 32);
      s += __shfl_xor(s, 2, 32);
      s += __shfl_xor(s, 4, 32);
      s += __shfl_xor(s, 8, 32);
      keep = (r == (p & 7)) ? s : keep;  // lane (p,hi) keeps point (p&7)+8*hi
    }
    if (p < 8) {
      out[outrow + m0 + p + 8 * hi] = sqrtf(fmaxf(keep, 1e-6f));
    }
  }
}

extern "C" void kernel_launch(void* const* d_in, const int* in_sizes, int n_in,
                              void* d_out, int out_size, void* d_ws, size_t ws_size,
                              hipStream_t stream) {
  const float* x1 = (const float*)d_in[0];
  const float* x2 = (const float*)d_in[1];
  const float* W1 = (const float*)d_in[2];
  const float* b1 = (const float*)d_in[3];
  const float* W2 = (const float*)d_in[4];
  const float* b2 = (const float*)d_in[5];
  float* out = (float*)d_out;

  dim3 block(kWaves * 32);
  dim3 grid((kWavesTotal + kWaves - 1) / kWaves);  // 1024 blocks x 8 waves
  hipLaunchKernelGGL(rmfd_wmma_kernel, grid, block, 0, stream,
                     x1, x2, W1, b1, W2, b2, out);
}
